// AttentionHead_40183714021696
// MI455X (gfx1250) — matrix operand order repaired and checked by reading, not compiled, not run on hardware
//
#include <hip/hip_runtime.h>
#include <math.h>
#include <type_traits>

// ---------------------------------------------------------------------------
// CDNA5 (gfx1250, wave32) WMMA bf16 attention head.
//   - matrix core : v_wmma_f32_16x16x32_bf16 (f32 accumulation)
//   - data mover  : tensor_load_to_lds (2D D#, padded LDS pitch), TENSORcnt,
//                   double-buffered LDS so the DMA overlaps the matrix core.
// ---------------------------------------------------------------------------

typedef __bf16   v16bf __attribute__((ext_vector_type(16)));
typedef float    v8f   __attribute__((ext_vector_type(8)));
typedef uint32_t u32x4 __attribute__((ext_vector_type(4)));
typedef uint32_t u32x8 __attribute__((ext_vector_type(8)));

union FragBF {
  v16bf v;
  uint4 q[2];
};
union G0 { u32x4 v; uint32_t w[4]; };
union G1 { u32x8 v; uint32_t w[8]; };

// ---------------------------------------------------------------------------
// TDM: DMA a (rows x 32) bf16 tile (memory row pitch = ld_elems) into LDS at
// lds_addr with an 80-byte padded LDS row pitch (pad 16B after every 64B).
//   group0: count=1 | lds_addr | global_addr | type=2
//   group1: data_size=2B, pad_enable, pad_interval=16DW, pad_amount=4DW,
//           tensor_dim0=tile_dim0=32, tensor_dim1=tile_dim1=rows,
//           tensor_dim0_stride=ld_elems
// 2D tensor -> two SGPR descriptor groups (VADDR2/3 = NULL).
// ---------------------------------------------------------------------------
__device__ __forceinline__ void tdm_load_tile_bf16(uint32_t lds_addr,
                                                   const __bf16* gptr,
                                                   uint32_t ld_elems,
                                                   uint32_t rows) {
  const uint64_t ga = (uint64_t)(uintptr_t)gptr;
  G0 g0;
  g0.w[0] = 1u;                    // [1:0] count=1 (valid), no gather
  g0.w[1] = lds_addr;              // [63:32] lds_addr (bytes)
  g0.w[2] = (uint32_t)ga;          // [95:64] global_addr[31:0]
  g0.w[3] = ((uint32_t)(ga >> 32) & 0x01FFFFFFu) | (2u << 30);  // addr[56:32]|type=2
  G1 g1;
  g1.w[0] = (1u << 16)             // data_size = 2 bytes
          | (1u << 20)             // pad_enable
          | (3u << 22)             // pad_interval: 16 DWORDs (64B)
          | (3u << 25);            // pad_amount : 4 DWORDs (16B)
  g1.w[1] = (32u << 16);           // tensor_dim0[15:0] = 32
  g1.w[2] = (rows << 16);          // tensor_dim0 hi=0 | tensor_dim1[15:0]=rows
  g1.w[3] = (32u << 16);           // tensor_dim1 hi=0 | tile_dim0=32
  g1.w[4] = rows;                  // tile_dim1=rows, tile_dim2=0
  g1.w[5] = ld_elems;              // tensor_dim0_stride[31:0]
  g1.w[6] = 0u;
  g1.w[7] = 0u;
  asm volatile("tensor_load_to_lds %0, %1" ::"s"(g0.v), "s"(g1.v) : "memory");
}

// Load 16 f32, convert to bf16 into a 16-byte-aligned local array.
__device__ __forceinline__ void load16_f32_to_bf16(const float* __restrict__ p,
                                                   __bf16* dst) {
  const float4* s = reinterpret_cast<const float4*>(p);
#pragma unroll
  for (int i = 0; i < 4; ++i) {
    float4 t = s[i];
    dst[4 * i + 0] = (__bf16)t.x;
    dst[4 * i + 1] = (__bf16)t.y;
    dst[4 * i + 2] = (__bf16)t.z;
    dst[4 * i + 3] = (__bf16)t.w;
  }
}

// ---------------------------------------------------------------------------
// Batched GEMM:  C[b] = alpha * A[b] x B'[b]^T     (B' stored [N x K] row-major)
//   AT = float  -> A converted f32->bf16 by VALU staging, B tile via TDM
//   AT = __bf16 -> both A and B tiles via TDM
//   TRANSC      -> store C transposed (C[n*ldc + m]) : used to emit V^T
// Block tile 256x128x32; 256 threads = 8 waves (4x2); wave tile 64x64 =
// 4x4 WMMA 16x16x32 tiles. Double-buffered LDS, one barrier per K-step.
// ---------------------------------------------------------------------------
template <typename AT, typename OT, bool TRANSC>
__global__ __launch_bounds__(256) void wmma_gemm(
    const AT* __restrict__ A, const __bf16* __restrict__ B, OT* __restrict__ C,
    int lda, int ldb, int ldc, int K,
    long long sA, long long sB, long long sC, float alpha) {
  constexpr int BM = 256, BN = 128, BK = 32, LDP = 40;  // 80B LDS row pitch
  constexpr bool A_F32 = std::is_same<AT, float>::value;
  __shared__ alignas(16) __bf16 As[2][BM][LDP];
  __shared__ alignas(16) __bf16 Bs[2][BN][LDP];  // B^T tile: [n][k]

  const AT*     Ab = A + (size_t)blockIdx.z * sA;
  const __bf16* Bb = B + (size_t)blockIdx.z * sB;
  OT*           Cb = C + (size_t)blockIdx.z * sC;

  const int n0 = blockIdx.x * BN;
  const int m0 = blockIdx.y * BM;

  const int tid  = threadIdx.x;
  const int lane = tid & 31;
  const int wid  = tid >> 5;
  const int wm   = wid >> 1;  // 0..3 -> 64 rows each
  const int wn   = wid & 1;   // 0..1 -> 64 cols each

  const int lrow    = lane & 15;
  const int halfsel = lane >> 4;

  v8f acc[4][4];
  const v8f vz = {0.f, 0.f, 0.f, 0.f, 0.f, 0.f, 0.f, 0.f};
#pragma unroll
  for (int i = 0; i < 4; ++i)
#pragma unroll
    for (int j = 0; j < 4; ++j) acc[i][j] = vz;

  const int T = K / BK;

  // ---- prologue: fill buffer 0 for k-step 0 ----
  if constexpr (A_F32) {
    if (wid == 0)
      tdm_load_tile_bf16((uint32_t)(uintptr_t)&Bs[0][0][0],
                         Bb + (size_t)n0 * ldb, (uint32_t)ldb, BN);
    {  // VALU-convert stage of the A tile: thread = one row of 32 elems
      alignas(16) __bf16 tmp[16];
      const float* src = (const float*)Ab + (size_t)(m0 + tid) * lda;
      load16_f32_to_bf16(src, tmp);
      *reinterpret_cast<uint4*>(&As[0][tid][0]) = reinterpret_cast<uint4*>(tmp)[0];
      *reinterpret_cast<uint4*>(&As[0][tid][8]) = reinterpret_cast<uint4*>(tmp)[1];
      load16_f32_to_bf16(src + 16, tmp);
      *reinterpret_cast<uint4*>(&As[0][tid][16]) = reinterpret_cast<uint4*>(tmp)[0];
      *reinterpret_cast<uint4*>(&As[0][tid][24]) = reinterpret_cast<uint4*>(tmp)[1];
    }
  } else {
    if (wid == 0) {
      tdm_load_tile_bf16((uint32_t)(uintptr_t)&As[0][0][0],
                         (const __bf16*)Ab + (size_t)m0 * lda, (uint32_t)lda, BM);
      tdm_load_tile_bf16((uint32_t)(uintptr_t)&Bs[0][0][0],
                         Bb + (size_t)n0 * ldb, (uint32_t)ldb, BN);
    }
  }

  for (int t = 0; t < T; ++t) {
    const int cur = t & 1;
    const int nxt = cur ^ 1;
    // fill(t) was the last TDM issued by this wave -> exact wait.
    if (wid == 0) __builtin_amdgcn_s_wait_tensorcnt(0);
    __syncthreads();  // buf[cur] ready for all; buf[nxt] reads (t-1) retired

    // ---- issue fill(t+1) into the alternate buffer; overlaps compute ----
    if (t + 1 < T) {
      const int k1 = (t + 1) * BK;
      if constexpr (A_F32) {
        if (wid == 0)
          tdm_load_tile_bf16((uint32_t)(uintptr_t)&Bs[nxt][0][0],
                             Bb + (size_t)n0 * ldb + k1, (uint32_t)ldb, BN);
        alignas(16) __bf16 tmp[16];
        const float* src = (const float*)Ab + (size_t)(m0 + tid) * lda + k1;
        load16_f32_to_bf16(src, tmp);
        *reinterpret_cast<uint4*>(&As[nxt][tid][0]) = reinterpret_cast<uint4*>(tmp)[0];
        *reinterpret_cast<uint4*>(&As[nxt][tid][8]) = reinterpret_cast<uint4*>(tmp)[1];
        load16_f32_to_bf16(src + 16, tmp);
        *reinterpret_cast<uint4*>(&As[nxt][tid][16]) = reinterpret_cast<uint4*>(tmp)[0];
        *reinterpret_cast<uint4*>(&As[nxt][tid][24]) = reinterpret_cast<uint4*>(tmp)[1];
      } else {
        if (wid == 0) {
          tdm_load_tile_bf16((uint32_t)(uintptr_t)&As[nxt][0][0],
                             (const __bf16*)Ab + (size_t)m0 * lda + k1, (uint32_t)lda, BM);
          tdm_load_tile_bf16((uint32_t)(uintptr_t)&Bs[nxt][0][0],
                             Bb + (size_t)n0 * ldb + k1, (uint32_t)ldb, BN);
        }
      }
    }

    // ---- compute on buf[cur]: 16 WMMAs ----
    // B fragments: lane%16 = column, lane/16 selects K half (0..15 / 16..31)
    FragBF fb[4];
#pragma unroll
    for (int j = 0; j < 4; ++j) {
      const __bf16* src = &Bs[cur][wn * 64 + j * 16 + lrow][halfsel * 16];
      fb[j].q[0] = reinterpret_cast<const uint4*>(src)[0];
      fb[j].q[1] = reinterpret_cast<const uint4*>(src)[1];
    }
#pragma unroll
    for (int i = 0; i < 4; ++i) {
      const __bf16* src = &As[cur][wm * 64 + i * 16 + lrow][halfsel * 16];
      FragBF fa;
      fa.q[0] = reinterpret_cast<const uint4*>(src)[0];       // K kb..kb+7
      fa.q[1] = reinterpret_cast<const uint4*>(src + 8)[0];   // K kb+8..kb+15
#pragma unroll
      for (int j = 0; j < 4; ++j) {
        acc[i][j] = __builtin_amdgcn_wmma_f32_16x16x32_bf16(
            false, fa.v, false, fb[j].v, (short)0, acc[i][j], false, false);
      }
    }
  }

  // ---- writeback: acc VGPR r -> row (r or r+8), col = lane%16 ----
#pragma unroll
  for (int i = 0; i < 4; ++i) {
    const int grow = m0 + wm * 64 + i * 16 + (halfsel ? 8 : 0);
#pragma unroll
    for (int j = 0; j < 4; ++j) {
      const int gcol = n0 + wn * 64 + j * 16 + lrow;
      if constexpr (TRANSC) {
        // 8 consecutive elements of one C^T row -> single 16B store (bf16)
        union { uint4 q; OT h[8]; } pk;
#pragma unroll
        for (int r = 0; r < 8; ++r) pk.h[r] = (OT)(acc[i][j][r] * alpha);
        *reinterpret_cast<uint4*>(Cb + (size_t)gcol * ldc + grow) = pk.q;
      } else {
#pragma unroll
        for (int r = 0; r < 8; ++r)
          Cb[(size_t)(grow + r) * ldc + gcol] = (OT)(acc[i][j][r] * alpha);
      }
    }
  }
}

// ---------------------------------------------------------------------------
// 32x32 tiled transpose with f32->bf16 conversion: Wt[n][k] = (bf16)W[k][n]
// ---------------------------------------------------------------------------
__global__ __launch_bounds__(256) void transpose_to_bf16(
    const float* __restrict__ W, __bf16* __restrict__ Wt, int rows, int cols) {
  __shared__ float tile[32][33];
  const int bx = blockIdx.x * 32;  // source col
  const int by = blockIdx.y * 32;  // source row
  const int tx = threadIdx.x & 31;
  const int ty = threadIdx.x >> 5;  // 8 rows per pass
#pragma unroll
  for (int r = ty; r < 32; r += 8)
    tile[r][tx] = W[(size_t)(by + r) * cols + bx + tx];
  __syncthreads();
#pragma unroll
  for (int r = ty; r < 32; r += 8)
    Wt[(size_t)(bx + r) * rows + by + tx] = (__bf16)tile[tx][r];
}

// ---------------------------------------------------------------------------
// Row softmax: one block per row; f32 scores in, bf16 probabilities out.
// ---------------------------------------------------------------------------
__global__ __launch_bounds__(256) void softmax_rows(
    const float* __restrict__ S, __bf16* __restrict__ P, int n) {
  const size_t row = blockIdx.x;
  const float* s = S + row * (size_t)n;
  __bf16*      p = P + row * (size_t)n;
  __shared__ float red[256];
  const int tid = threadIdx.x;

  float lm = -INFINITY;
  for (int i = tid; i < n; i += 256) lm = fmaxf(lm, s[i]);
  red[tid] = lm;
  __syncthreads();
  for (int off = 128; off > 0; off >>= 1) {
    if (tid < off) red[tid] = fmaxf(red[tid], red[tid + off]);
    __syncthreads();
  }
  const float m = red[0];
  __syncthreads();

  float ls = 0.f;
  for (int i = tid; i < n; i += 256) ls += __expf(s[i] - m);
  red[tid] = ls;
  __syncthreads();
  for (int off = 128; off > 0; off >>= 1) {
    if (tid < off) red[tid] += red[tid + off];
    __syncthreads();
  }
  const float inv = 1.0f / red[0];

  for (int i = tid; i < n; i += 256) p[i] = (__bf16)(__expf(s[i] - m) * inv);
}

// ---------------------------------------------------------------------------
// Host-side orchestration.
// ---------------------------------------------------------------------------
extern "C" void kernel_launch(void* const* d_in, const int* in_sizes, int n_in,
                              void* d_out, int out_size, void* d_ws, size_t ws_size,
                              hipStream_t stream) {
  constexpr int B = 4, S = 2048, E = 1024;
  constexpr int BS = B * S;  // 8192

  const float* Xk = (const float*)d_in[0];
  const float* Xv = (const float*)d_in[1];
  const float* Xq = (const float*)d_in[2];
  const float* WK = (const float*)d_in[3];
  const float* WV = (const float*)d_in[4];
  const float* WQ = (const float*)d_in[5];
  float* out = (float*)d_out;

  char* ws = (char*)d_ws;
  const size_t qkvBytes = (size_t)BS * E * sizeof(__bf16);    // 16 MB each
  const size_t scBytes  = (size_t)B * S * S * sizeof(float);  // 64 MB
  const size_t pBytes   = (size_t)B * S * S * sizeof(__bf16); // 32 MB
  const size_t wBytes   = (size_t)E * E * sizeof(__bf16);     // 2 MB each
  __bf16* Q   = (__bf16*)(ws);
  __bf16* Kb  = (__bf16*)(ws + qkvBytes);
  __bf16* Vt  = (__bf16*)(ws + 2 * qkvBytes);                 // [E][BS] = V^T
  float*  SC  = (float*) (ws + 3 * qkvBytes);
  __bf16* P   = (__bf16*)(ws + 3 * qkvBytes + scBytes);
  __bf16* WKt = (__bf16*)(ws + 3 * qkvBytes + scBytes + pBytes);
  __bf16* WVt = (__bf16*)((char*)WKt + wBytes);
  __bf16* WQt = (__bf16*)((char*)WKt + 2 * wBytes);

  const dim3 blk(256);
  const float one = 1.0f;
  const float inv_sqrt_s = 1.0f / sqrtf((float)S);  // reference: sqrt(seq_len)

  // --- Weights -> bf16, transposed to [N][K] so every GEMM consumes B rows
  {
    dim3 grid(E / 32, E / 32);
    transpose_to_bf16<<<grid, blk, 0, stream>>>(WK, WKt, E, E);
    transpose_to_bf16<<<grid, blk, 0, stream>>>(WV, WVt, E, E);
    transpose_to_bf16<<<grid, blk, 0, stream>>>(WQ, WQt, E, E);
  }

  // --- Projections: [8192x1024] x W -> bf16.  V is written transposed.
  {
    dim3 grid(E / 128, BS / 256, 1);
    wmma_gemm<float, __bf16, false><<<grid, blk, 0, stream>>>(
        Xk, WKt, Kb, E, E, E, E, 0LL, 0LL, 0LL, one);
    wmma_gemm<float, __bf16, true><<<grid, blk, 0, stream>>>(
        Xv, WVt, Vt, E, E, BS, E, 0LL, 0LL, 0LL, one);  // Vt: [E][BS], ldc=BS
    wmma_gemm<float, __bf16, false><<<grid, blk, 0, stream>>>(
        Xq, WQt, Q, E, E, E, E, 0LL, 0LL, 0LL, one);
  }

  // --- Scores: per batch, Q[2048x1024] x K^T -> f32, scaled by 1/sqrt(S)
  {
    dim3 grid(S / 128, S / 256, B);
    wmma_gemm<__bf16, float, false><<<grid, blk, 0, stream>>>(
        Q, Kb, SC, E, E, S, E,
        (long long)S * E, (long long)S * E, (long long)S * S, inv_sqrt_s);
  }

  // --- Softmax over each of B*S rows of length S; write bf16 P
  softmax_rows<<<dim3(B * S), blk, 0, stream>>>(SC, P, S);

  // --- Output: per batch, P[2048x2048] x (V^T)^T -> f32 d_out
  {
    dim3 grid(E / 128, S / 256, B);
    wmma_gemm<__bf16, float, false><<<grid, blk, 0, stream>>>(
        P, Vt, out, S, BS, E, S,
        (long long)S * S, (long long)S, (long long)S * E, one);
  }
}
